// ArcDecoderLayer_35287451304110
// MI455X (gfx1250) — matrix operand order repaired and checked by reading, not compiled, not run on hardware
//
#include <hip/hip_runtime.h>

// ---------------------------------------------------------------------------
// Problem constants (from reference)
// ---------------------------------------------------------------------------
constexpr int Sq = 2048;   // sequence length
constexpr int Dm = 2048;   // model dim
constexpr int FF = 8192;   // ffn dim
constexpr int NH = 32;     // heads
constexpr int HD = 64;     // head dim
constexpr int LL = 4096;   // 2*S (mem ++ hidden)
constexpr float EPSV = 1e-5f;
#define NEG_BIG (-3.0e38f)

typedef __bf16 bf16;
typedef __attribute__((ext_vector_type(16))) __bf16 v16bf;
typedef __attribute__((ext_vector_type(8)))  float  v8f;
typedef __attribute__((ext_vector_type(4)))  unsigned int v4u;
typedef __attribute__((ext_vector_type(8)))  int v8i;
typedef __attribute__((ext_vector_type(4)))  int v4i;

#if defined(__has_builtin)
#if __has_builtin(__builtin_amdgcn_tensor_load_to_lds) && \
    __has_builtin(__builtin_amdgcn_s_wait_tensorcnt)
#define HAVE_TDM 1
#endif
#endif

// ---------------------------------------------------------------------------
// WMMA helpers (CDNA5 16x16x32 bf16, wave32)
// A 16x32 frag: lane (m = lane&15, hf = lane>>4): elems 0..15 -> K = hf*16 + 0..15
// B 32x16 frag: lane (n = lane&15, hf): elems 0..15 -> K = hf*16 + 0..15
// C/D 16x16:    VGPR r, lane: row = r + 8*hf, col = lane&15
// ---------------------------------------------------------------------------
__device__ __forceinline__ v8f wmma_bf16(v16bf a, v16bf b, v8f c) {
  return __builtin_amdgcn_wmma_f32_16x16x32_bf16(false, a, false, b, (short)0, c,
                                                 false, false);
}

__device__ __forceinline__ v16bf load_frag_a(const bf16* base, int ld, int m,
                                             int koff, int hf) {
  v16bf a;
  const bf16* p = base + (size_t)m * ld + koff + hf * 16;
  #pragma unroll
  for (int i = 0; i < 16; ++i) a[i] = p[i];
  return a;
}

__device__ __forceinline__ v16bf load_frag_b(const bf16* base, int ld, int n,
                                             int koff, int hf) {
  v16bf b;
  const bf16* p = base + (size_t)n * ld + koff + hf * 16;
  #pragma unroll
  for (int i = 0; i < 16; ++i) b[i] = p[i];
  return b;
}

// ---------------------------------------------------------------------------
// TDM: DMA one 128-row x 32-element (bf16) tile, global -> LDS, packed
// row-major into LDS (row stride 32 elems). D# layout per ISA 08 §8.3/8.4.
// Issued by a single wave; completion via TENSORcnt.
// Toolchain here is the 6-arg builtin variant (clang-23 / therock-10.0):
//   (uint32x4 g0, int32x8 g1, int32x4 g2, int32x4 g3, int32x8, i32 cpol)
// ---------------------------------------------------------------------------
__device__ __forceinline__ void tdm_load_tile_128x32(unsigned int lds_addr,
                                                     unsigned long long gaddr,
                                                     int K) {
#ifdef HAVE_TDM
  v4u g0;
  g0[0] = 1u;                                   // count=1, user descriptor
  g0[1] = lds_addr;                             // LDS byte address
  g0[2] = (unsigned int)gaddr;                  // global_addr[31:0]
  g0[3] = (unsigned int)((gaddr >> 32) & 0x1FFFFFFull) | (2u << 30);  // type=2
  v8i g1;
  g1[0] = (int)(1u << 16);                      // data_size=1 (2 bytes)
  g1[1] = (int)((unsigned)(K & 0xFFFF) << 16);  // tensor_dim0 low16 @ [63:48]
  g1[2] = (int)(((unsigned)K >> 16) | (128u << 16));  // dim0 hi | tensor_dim1
  g1[3] = (int)(32u << 16);                     // tile_dim0=32 @ [127:112]
  g1[4] = 128;                                  // tile_dim1=128
  g1[5] = K;                                    // tensor_dim0_stride low 32
  g1[6] = 0;
  g1[7] = 0;
  v4i z4 = {};
  v8i z8 = {};
  __builtin_amdgcn_tensor_load_to_lds(g0, g1, z4, z4, z8, 0);
#else
  (void)lds_addr; (void)gaddr; (void)K;
#endif
}

// ---------------------------------------------------------------------------
// Elementwise / normalization kernels
// ---------------------------------------------------------------------------
__global__ void cvt_bf16_kernel(const float* __restrict__ src,
                                bf16* __restrict__ dst, size_t n) {
  size_t i = (size_t)blockIdx.x * blockDim.x + threadIdx.x;
  if (i < n) dst[i] = (bf16)src[i];
}

__global__ __launch_bounds__(256) void ln_kernel(const float* __restrict__ src,
                                                 const float* __restrict__ w,
                                                 const float* __restrict__ b,
                                                 bf16* __restrict__ out) {
  const int row = blockIdx.x;
  const float* x = src + (size_t)row * Dm;
  bf16* y = out + (size_t)row * Dm;
  float s = 0.f, s2 = 0.f;
  for (int i = threadIdx.x; i < Dm; i += 256) {
    float v = x[i];
    s += v;
    s2 += v * v;
  }
  __shared__ float rs[256], rs2[256];
  rs[threadIdx.x] = s;
  rs2[threadIdx.x] = s2;
  __syncthreads();
  for (int st = 128; st > 0; st >>= 1) {
    if (threadIdx.x < st) {
      rs[threadIdx.x] += rs[threadIdx.x + st];
      rs2[threadIdx.x] += rs2[threadIdx.x + st];
    }
    __syncthreads();
  }
  const float mean = rs[0] / (float)Dm;
  const float var = rs2[0] / (float)Dm - mean * mean;
  const float inv = rsqrtf(var + EPSV);
  for (int i = threadIdx.x; i < Dm; i += 256)
    y[i] = (bf16)((x[i] - mean) * inv * w[i] + b[i]);
}

// rope: src f32 [rows][Dm] -> dst bf16 [NH][rows][HD], rotary on first 16 dims
__global__ void rope_kernel(const float* __restrict__ src,
                            const int* __restrict__ pos_ids,
                            bf16* __restrict__ dst, int rows) {
  int idx = blockIdx.x * blockDim.x + threadIdx.x;
  if (idx >= rows * NH) return;
  const int j = idx / NH, hh = idx % NH;
  const float* x = src + (size_t)j * Dm + hh * HD;
  bf16* y = dst + ((size_t)hh * rows + j) * HD;
  const float t = (float)pos_ids[j % Sq];
  #pragma unroll
  for (int i = 0; i < 8; ++i) {
    float f = __powf(10000.0f, -(float)(2 * i) / 16.0f);
    float c = __cosf(t * f), s = __sinf(t * f);
    float x1 = x[i], x2 = x[i + 8];
    y[i]     = (bf16)(x1 * c - x2 * s);
    y[i + 8] = (bf16)(x2 * c + x1 * s);
  }
  #pragma unroll
  for (int i = 16; i < HD; ++i) y[i] = (bf16)x[i];
}

// v transpose: f32 [LL][Dm] -> bf16 [NH][HD][LL]
__global__ void vtrans_kernel(const float* __restrict__ vf,
                              bf16* __restrict__ vt) {
  int idx = blockIdx.x * blockDim.x + threadIdx.x;
  if (idx >= LL * NH) return;
  const int j = idx / NH, hh = idx % NH;
  #pragma unroll
  for (int d = 0; d < HD; ++d)
    vt[((size_t)hh * HD + d) * LL + j] = (bf16)vf[(size_t)j * Dm + hh * HD + d];
}

__global__ void silu_mul_kernel(const float* __restrict__ g,
                                const float* __restrict__ u,
                                bf16* __restrict__ act, size_t n) {
  size_t i = (size_t)blockIdx.x * blockDim.x + threadIdx.x;
  if (i >= n) return;
  float gv = g[i];
  float s = gv / (1.f + __expf(-gv));
  act[i] = (bf16)(s * u[i]);
}

// ---------------------------------------------------------------------------
// Tiled WMMA GEMM: C[M][N] (f32) = A[M][K](bf16) @ W[N][K](bf16)^T (+ res)
// Block 128x128, BK=32, 256 threads = 8 waves; wave = 32x64 patch (8 tiles).
// Tile staging: Tensor Data Mover (wave 0 issues 2 descriptors / iteration),
// falling back to vector loads if the TDM builtin is unavailable.
// ---------------------------------------------------------------------------
template <bool ADD_RES>
__global__ __launch_bounds__(256) void gemm_kernel(
    const bf16* __restrict__ A, const bf16* __restrict__ W,
    const float* __restrict__ res, float* __restrict__ C, int M, int N, int K) {
  __shared__ bf16 sA[128 * 32];
  __shared__ bf16 sW[128 * 32];

  const int tid = threadIdx.x;
  const int lane = tid & 31, wv = tid >> 5;
  const int n16 = lane & 15, hf = lane >> 4;
  const int wr = wv & 3, wc = wv >> 2;           // 4 row-waves x 2 col-waves
  const int m0 = blockIdx.y * 128, n0 = blockIdx.x * 128;

#ifdef HAVE_TDM
  const unsigned int lds_a = (unsigned int)(uintptr_t)sA;
  const unsigned int lds_w = (unsigned int)(uintptr_t)sW;
#endif

  v8f acc[2][4] = {};

  for (int k0 = 0; k0 < K; k0 += 32) {
#ifdef HAVE_TDM
    if (wv == 0) {
      tdm_load_tile_128x32(
          lds_a, (unsigned long long)(uintptr_t)(A + (size_t)m0 * K + k0), K);
      tdm_load_tile_128x32(
          lds_w, (unsigned long long)(uintptr_t)(W + (size_t)n0 * K + k0), K);
      __builtin_amdgcn_s_wait_tensorcnt(0);
    }
#else
    #pragma unroll
    for (int it = 0; it < 2; ++it) {
      int idx = tid + it * 256;                  // 512 chunks of 8 bf16
      int r = idx >> 2, c = (idx & 3) << 3;
      *(uint4*)(sA + r * 32 + c) =
          *(const uint4*)(A + (size_t)(m0 + r) * K + k0 + c);
      *(uint4*)(sW + r * 32 + c) =
          *(const uint4*)(W + (size_t)(n0 + r) * K + k0 + c);
    }
#endif
    __syncthreads();

    v16bf af[2];
    #pragma unroll
    for (int mt = 0; mt < 2; ++mt)
      af[mt] = load_frag_a(sA, 32, wr * 32 + mt * 16 + n16, 0, hf);
    #pragma unroll
    for (int nt = 0; nt < 4; ++nt) {
      v16bf bf_ = load_frag_b(sW, 32, wc * 64 + nt * 16 + n16, 0, hf);
      #pragma unroll
      for (int mt = 0; mt < 2; ++mt)
        acc[mt][nt] = wmma_bf16(af[mt], bf_, acc[mt][nt]);
    }
    __syncthreads();
  }

  #pragma unroll
  for (int mt = 0; mt < 2; ++mt)
    #pragma unroll
    for (int nt = 0; nt < 4; ++nt)
      #pragma unroll
      for (int r = 0; r < 8; ++r) {
        int gm = m0 + wr * 32 + mt * 16 + r + hf * 8;
        int gn = n0 + wc * 64 + nt * 16 + n16;
        float v = acc[mt][nt][r];
        if (ADD_RES) v += res[(size_t)gm * N + gn];
        C[(size_t)gm * N + gn] = v;
      }
}

// ---------------------------------------------------------------------------
// Flash attention, WMMA, online softmax.
// grid = (NH, Sq/128), block = 256 (8 waves). Each wave: one 16-row q tile.
// Mask (derived from reference): query m attends memory keys j<m and the
// single hidden key j==S+m.
// ---------------------------------------------------------------------------
__global__ __launch_bounds__(256) void attn_kernel(
    const bf16* __restrict__ qh,   // [NH][Sq][HD]
    const bf16* __restrict__ kh,   // [NH][LL][HD]
    const bf16* __restrict__ vt,   // [NH][HD][LL]
    bf16* __restrict__ attn) {     // [Sq][Dm]
  __shared__ bf16 P[8][16 * 32];

  const int tid = threadIdx.x;
  const int lane = tid & 31, wv = tid >> 5;
  const int n16 = lane & 15, hf = lane >> 4;
  const int head = blockIdx.x;
  const int qbase = (blockIdx.y * 8 + wv) * 16;

  const bf16* qb = qh + ((size_t)head * Sq + qbase) * HD;
  const bf16* kb = kh + (size_t)head * LL * HD;
  const bf16* vb = vt + (size_t)head * HD * LL;

  const v16bf qa0 = load_frag_a(qb, HD, n16, 0, hf);
  const v16bf qa1 = load_frag_a(qb, HD, n16, 32, hf);

  v8f o[4] = {};
  float mrow[8], lrow[8];
  #pragma unroll
  for (int r = 0; r < 8; ++r) { mrow[r] = NEG_BIG; lrow[r] = 0.f; }

  bf16* Pw = P[wv];

  for (int kb0 = 0; kb0 < LL; kb0 += 32) {
    // scores for key sub-tiles t=0 (keys kb0+n) and t=1 (keys kb0+16+n)
    v8f sc0 = {}, sc1 = {};
    {
      v16bf b00 = load_frag_b(kb + (size_t)kb0 * HD, HD, n16, 0, hf);
      v16bf b01 = load_frag_b(kb + (size_t)kb0 * HD, HD, n16, 32, hf);
      sc0 = wmma_bf16(qa0, b00, sc0);
      sc0 = wmma_bf16(qa1, b01, sc0);
      v16bf b10 = load_frag_b(kb + (size_t)(kb0 + 16) * HD, HD, n16, 0, hf);
      v16bf b11 = load_frag_b(kb + (size_t)(kb0 + 16) * HD, HD, n16, 32, hf);
      sc1 = wmma_bf16(qa0, b10, sc1);
      sc1 = wmma_bf16(qa1, b11, sc1);
    }

    const int j0 = kb0 + n16, j1 = kb0 + 16 + n16;
    float alpha[8];
    #pragma unroll
    for (int r = 0; r < 8; ++r) {
      const int mg = qbase + r + hf * 8;
      const bool a0 = (j0 < Sq) ? (j0 < mg) : ((j0 - Sq) == mg);
      const bool a1 = (j1 < Sq) ? (j1 < mg) : ((j1 - Sq) == mg);
      float s0 = sc0[r] * 0.125f + (a0 ? 0.f : NEG_BIG);
      float s1 = sc1[r] * 0.125f + (a1 ? 0.f : NEG_BIG);
      float rm = fmaxf(s0, s1);
      #pragma unroll
      for (int off = 1; off < 16; off <<= 1)
        rm = fmaxf(rm, __shfl_xor(rm, off, 32));
      const float mnew = fmaxf(mrow[r], rm);
      alpha[r] = __expf(mrow[r] - mnew);
      float p0 = __expf(s0 - mnew);
      float p1 = __expf(s1 - mnew);
      Pw[(r + hf * 8) * 32 + n16] = (bf16)p0;
      Pw[(r + hf * 8) * 32 + 16 + n16] = (bf16)p1;
      float rsum = p0 + p1;
      #pragma unroll
      for (int off = 1; off < 16; off <<= 1) rsum += __shfl_xor(rsum, off, 32);
      lrow[r] = lrow[r] * alpha[r] + rsum;
      mrow[r] = mnew;
    }
    #pragma unroll
    for (int nt = 0; nt < 4; ++nt)
      #pragma unroll
      for (int r = 0; r < 8; ++r) o[nt][r] *= alpha[r];

    // P written to LDS above is gathered cross-lane below: LDS ops from one
    // wave are in-order; the wait + memory clobber keeps the compiler honest.
    asm volatile("s_wait_dscnt 0" ::: "memory");

    const v16bf pa = load_frag_a(Pw, 32, n16, 0, hf);
    #pragma unroll
    for (int nt = 0; nt < 4; ++nt) {
      v16bf vfr = load_frag_b(vb + kb0, LL, nt * 16 + n16, 0, hf);
      o[nt] = wmma_bf16(pa, vfr, o[nt]);
    }
  }

  #pragma unroll
  for (int nt = 0; nt < 4; ++nt)
    #pragma unroll
    for (int r = 0; r < 8; ++r) {
      int gm = qbase + r + hf * 8;
      int gn = head * HD + nt * 16 + n16;
      attn[(size_t)gm * Dm + gn] = (bf16)(o[nt][r] / lrow[r]);
    }
}

// ---------------------------------------------------------------------------
// Host orchestration
// ---------------------------------------------------------------------------
extern "C" void kernel_launch(void* const* d_in, const int* in_sizes, int n_in,
                              void* d_out, int out_size, void* d_ws,
                              size_t ws_size, hipStream_t stream) {
  (void)in_sizes; (void)n_in; (void)out_size; (void)ws_size;
  const float* hidden = (const float*)d_in[0];
  const float* memory = (const float*)d_in[1];
  // d_in[2] attention_mask: causal; mask is applied analytically in-kernel.
  const int* pos = (const int*)d_in[3];
  const float* ln1w = (const float*)d_in[4];
  const float* ln1b = (const float*)d_in[5];
  const float* ln2w = (const float*)d_in[6];
  const float* ln2b = (const float*)d_in[7];
  const float* Wq = (const float*)d_in[8];
  const float* Wk = (const float*)d_in[9];
  const float* Wv = (const float*)d_in[10];
  const float* Wo = (const float*)d_in[11];
  const float* Wg = (const float*)d_in[12];
  const float* Wu = (const float*)d_in[13];
  const float* Wd = (const float*)d_in[14];

  char* base = (char*)d_ws;
  size_t off = 0;
  auto alloc = [&](size_t bytes) -> void* {
    void* r = base + off;
    off = (off + bytes + 255) & ~(size_t)255;
    return r;
  };

  bf16* xb  = (bf16*)alloc((size_t)LL * Dm * 2);
  bf16* wqb = (bf16*)alloc((size_t)Dm * Dm * 2);
  bf16* wkb = (bf16*)alloc((size_t)Dm * Dm * 2);
  bf16* wvb = (bf16*)alloc((size_t)Dm * Dm * 2);
  bf16* wob = (bf16*)alloc((size_t)Dm * Dm * 2);
  bf16* wgb = (bf16*)alloc((size_t)FF * Dm * 2);
  bf16* wub = (bf16*)alloc((size_t)FF * Dm * 2);
  bf16* wdb = (bf16*)alloc((size_t)Dm * FF * 2);
  float* qf = (float*)alloc((size_t)Sq * Dm * 4);
  float* kf = (float*)alloc((size_t)LL * Dm * 4);
  float* vf = (float*)alloc((size_t)LL * Dm * 4);
  bf16* qhB = (bf16*)alloc((size_t)NH * Sq * HD * 2);
  bf16* khB = (bf16*)alloc((size_t)NH * LL * HD * 2);
  bf16* vtB = (bf16*)alloc((size_t)NH * HD * LL * 2);
  bf16* attnB = (bf16*)alloc((size_t)Sq * Dm * 2);
  float* h1 = (float*)alloc((size_t)Sq * Dm * 4);
  bf16* h2 = (bf16*)alloc((size_t)Sq * Dm * 2);
  float* gbuf = (float*)alloc((size_t)Sq * FF * 4);
  float* ubuf = (float*)alloc((size_t)Sq * FF * 4);
  bf16* act = (bf16*)alloc((size_t)Sq * FF * 2);

  auto cvt = [&](const float* s, bf16* d, size_t n) {
    cvt_bf16_kernel<<<(unsigned)((n + 255) / 256), 256, 0, stream>>>(s, d, n);
  };
  cvt(Wq, wqb, (size_t)Dm * Dm);
  cvt(Wk, wkb, (size_t)Dm * Dm);
  cvt(Wv, wvb, (size_t)Dm * Dm);
  cvt(Wo, wob, (size_t)Dm * Dm);
  cvt(Wg, wgb, (size_t)FF * Dm);
  cvt(Wu, wub, (size_t)FF * Dm);
  cvt(Wd, wdb, (size_t)Dm * FF);

  // LN1 -> x = [ LN(memory) ; LN(hidden) ] (bf16)
  ln_kernel<<<Sq, 256, 0, stream>>>(memory, ln1w, ln1b, xb);
  ln_kernel<<<Sq, 256, 0, stream>>>(hidden, ln1w, ln1b, xb + (size_t)Sq * Dm);

  // Q (hidden rows only), K, V projections
  gemm_kernel<false><<<dim3(Dm / 128, Sq / 128), 256, 0, stream>>>(
      xb + (size_t)Sq * Dm, wqb, nullptr, qf, Sq, Dm, Dm);
  gemm_kernel<false><<<dim3(Dm / 128, LL / 128), 256, 0, stream>>>(
      xb, wkb, nullptr, kf, LL, Dm, Dm);
  gemm_kernel<false><<<dim3(Dm / 128, LL / 128), 256, 0, stream>>>(
      xb, wvb, nullptr, vf, LL, Dm, Dm);

  rope_kernel<<<(Sq * NH + 255) / 256, 256, 0, stream>>>(qf, pos, qhB, Sq);
  rope_kernel<<<(LL * NH + 255) / 256, 256, 0, stream>>>(kf, pos, khB, LL);
  vtrans_kernel<<<(LL * NH + 255) / 256, 256, 0, stream>>>(vf, vtB);

  attn_kernel<<<dim3(NH, Sq / 128), 256, 0, stream>>>(qhB, khB, vtB, attnB);

  // O projection + residual
  gemm_kernel<true><<<dim3(Dm / 128, Sq / 128), 256, 0, stream>>>(
      attnB, wob, hidden, h1, Sq, Dm, Dm);

  // LN2 + MLP
  ln_kernel<<<Sq, 256, 0, stream>>>(h1, ln2w, ln2b, h2);
  gemm_kernel<false><<<dim3(FF / 128, Sq / 128), 256, 0, stream>>>(
      h2, wgb, nullptr, gbuf, Sq, FF, Dm);
  gemm_kernel<false><<<dim3(FF / 128, Sq / 128), 256, 0, stream>>>(
      h2, wub, nullptr, ubuf, Sq, FF, Dm);
  silu_mul_kernel<<<(unsigned)(((size_t)Sq * FF + 255) / 256), 256, 0, stream>>>(
      gbuf, ubuf, act, (size_t)Sq * FF);
  gemm_kernel<true><<<dim3(Dm / 128, Sq / 128), 256, 0, stream>>>(
      act, wdb, h1, (float*)d_out, Sq, Dm, FF);
}
